// WireframeGNNHead_10943576671012
// MI455X (gfx1250) — compile-verified, matrix operand order checked by reading, not run on hardware
//
#include <hip/hip_runtime.h>

// ---------------------------------------------------------------------------
// Types / helpers for CDNA5 WMMA
// ---------------------------------------------------------------------------
typedef __attribute__((ext_vector_type(16))) __bf16 v16bf;
typedef __attribute__((ext_vector_type(8)))  float  v8f;
typedef int gv4i __attribute__((vector_size(16)));   // GCC vector: matches builtin

#define AS1 __attribute__((address_space(1)))
#define AS3 __attribute__((address_space(3)))

__device__ __forceinline__ unsigned short f2bf(float x) {
    unsigned int u = __builtin_bit_cast(unsigned int, x);
    u += 0x7FFFu + ((u >> 16) & 1u);          // round-to-nearest-even
    return (unsigned short)(u >> 16);
}
__device__ __forceinline__ float bf2f(unsigned short h) {
    unsigned int u = (unsigned int)h << 16;
    return __builtin_bit_cast(float, u);
}

#define LDS_STRIDE 20   // uints per row: 80B, 16B-aligned, non-pow2 (bank spread)

// Async load-to-LDS path (gfx1250 ASYNCcnt DMA) when the toolchain exposes it.
#ifdef __has_builtin
#  if __has_builtin(__builtin_amdgcn_global_load_async_to_lds_b128)
#    define USE_ASYNC_LDS 1
#  endif
#endif
#ifndef USE_ASYNC_LDS
#  define USE_ASYNC_LDS 0
#endif

#if USE_ASYNC_LDS
__device__ __forceinline__ void wait_async0() {
#  if __has_builtin(__builtin_amdgcn_s_wait_asynccnt)
    __builtin_amdgcn_s_wait_asynccnt(0);
#  else
    asm volatile("s_wait_asynccnt 0x0" ::: "memory");
#  endif
}
#endif

// ---------------------------------------------------------------------------
// Tiled bf16 WMMA GEMM:  C[r, cOff+c] = act( A[r,:] @ W[:,c] + bias[c] )
// A:  bf16 [N x K], element row stride lda (may be a concat-buffer slice)
// Wt: bf16 [M x K] (pre-transposed weights), row stride ldw == K
// Cout: fp32 or bf16 (out_bf16), element row stride ldc, column offset cOff
// 128x64 tile / 256-thread block; 8 wave32 waves x (2x2) 16x16x32 fragments.
// Double-buffered LDS, one barrier per K-step; staging is a pure byte copy
// (async load-to-LDS when available).
// Requires K % 32 == 0, M % 64 == 0.
// ---------------------------------------------------------------------------
__launch_bounds__(256)
__global__ void gemm_wmma(const unsigned short* __restrict__ A, int lda,
                          const unsigned short* __restrict__ Wt, int ldw,
                          const float* __restrict__ bias,
                          void* __restrict__ Cout, int ldc, int cOff, int out_bf16,
                          int Nrows, int Kdim, int relu)
{
    constexpr int STR  = LDS_STRIDE;
    constexpr int ABUF = 128 * STR;
    constexpr int BBUF = 64 * STR;
    __shared__ unsigned int lA[2 * ABUF];   // [buf][row][k-pair] packed bf16x2
    __shared__ unsigned int lB[2 * BBUF];   // [buf][n]  [k-pair]

    const int t    = threadIdx.x;
    const int lane = t & 31;
    const int w    = t >> 5;
    const int wr   = w >> 1;               // wave row 0..3 (32 rows each)
    const int wc   = w & 1;                // wave col 0..1 (32 cols each)
    const int bm   = blockIdx.y * 128;
    const int bn   = blockIdx.x * 64;
    const int half = lane >> 4;
    const int mrow = lane & 15;

    // staging: A row = t>>1, two 16B chunks; B n-row = t>>2, one 16B chunk
    const int a_row = t >> 1;
    const int a_c   = (t & 1) * 2;         // uint4 chunk base (0|2)
    const int b_n   = t >> 2;
    const int b_c   = t & 3;

    // clamp: OOB rows stage valid garbage that lands only in unsaved C rows
    const int    a_gr  = min(bm + a_row, Nrows - 1);
    const size_t aBase = (size_t)a_gr * lda;
    const size_t bBase = (size_t)(bn + b_n) * ldw;

    union Frag { v16bf v; uint4 q[2]; };
    v8f acc[2][2] = {};

    auto compute = [&](int buf) {
        Frag fa[2], fb[2];
        const unsigned int* baseA = &lA[buf * ABUF];
        const unsigned int* baseB = &lB[buf * BBUF];
        #pragma unroll
        for (int fr = 0; fr < 2; ++fr) {
            const unsigned int* p = baseA + (wr * 32 + fr * 16 + mrow) * STR + half * 4;
            fa[fr].q[0] = *(const uint4*)p;
            fa[fr].q[1] = *(const uint4*)(p + 8);
        }
        #pragma unroll
        for (int fc = 0; fc < 2; ++fc) {
            const unsigned int* p = baseB + (wc * 32 + fc * 16 + mrow) * STR + half * 4;
            fb[fc].q[0] = *(const uint4*)p;
            fb[fc].q[1] = *(const uint4*)(p + 8);
        }
        #pragma unroll
        for (int fr = 0; fr < 2; ++fr)
            #pragma unroll
            for (int fc = 0; fc < 2; ++fc)
                acc[fr][fc] = __builtin_amdgcn_wmma_f32_16x16x32_bf16(
                    false, fa[fr].v, false, fb[fc].v,
                    (short)0, acc[fr][fc], false, false);
    };

    const int nk = Kdim >> 5;
    int cur = 0;

#if USE_ASYNC_LDS
    auto stage = [&](int buf, int k0) {
        const unsigned short* ga = A + aBase + k0 + a_c * 8;
        unsigned int*         la = &lA[buf * ABUF + a_row * STR + a_c * 4];
        __builtin_amdgcn_global_load_async_to_lds_b128(
            (AS1 gv4i*)ga,       (AS3 gv4i*)la,       0, 0);
        __builtin_amdgcn_global_load_async_to_lds_b128(
            (AS1 gv4i*)(ga + 8), (AS3 gv4i*)(la + 4), 0, 0);
        const unsigned short* gb = Wt + bBase + k0 + b_c * 8;
        unsigned int*         lb = &lB[buf * BBUF + b_n * STR + b_c * 4];
        __builtin_amdgcn_global_load_async_to_lds_b128(
            (AS1 gv4i*)gb,       (AS3 gv4i*)lb,       0, 0);
    };
    stage(0, 0);
    for (int i = 0; i < nk; ++i) {
        wait_async0();                       // own async writes done
        __syncthreads();                     // everyone's writes visible
        if (i + 1 < nk) stage(cur ^ 1, (i + 1) * 32);
        compute(cur);
        cur ^= 1;
    }
#else
    uint4 ga0, ga1, gb;
    auto loadr = [&](int k0) {
        const uint4* pa = (const uint4*)(A + aBase + k0 + a_c * 8);
        ga0 = pa[0]; ga1 = pa[1];
        gb  = *(const uint4*)(Wt + bBase + k0 + b_c * 8);
    };
    auto storer = [&](int buf) {
        uint4* da = (uint4*)&lA[buf * ABUF + a_row * STR + a_c * 4];
        da[0] = ga0; da[1] = ga1;
        *(uint4*)&lB[buf * BBUF + b_n * STR + b_c * 4] = gb;
    };
    loadr(0); storer(0);
    for (int i = 0; i < nk; ++i) {
        if (i + 1 < nk) loadr((i + 1) * 32); // global latency overlaps compute
        __syncthreads();
        compute(cur);
        if (i + 1 < nk) storer(cur ^ 1);     // other buffer: no barrier needed
        cur ^= 1;
    }
#endif

    // ---- epilogue: bias + optional ReLU, fp32 or bf16 strided store ----
    const int colBase = bn + wc * 32 + mrow;
    const float b0 = bias ? bias[colBase]      : 0.0f;
    const float b1 = bias ? bias[colBase + 16] : 0.0f;
    #pragma unroll
    for (int fr = 0; fr < 2; ++fr) {
        #pragma unroll
        for (int r = 0; r < 8; ++r) {
            const int row = bm + wr * 32 + fr * 16 + half * 8 + r;
            if (row < Nrows) {
                float v0 = acc[fr][0][r] + b0;
                float v1 = acc[fr][1][r] + b1;
                if (relu) { v0 = fmaxf(v0, 0.0f); v1 = fmaxf(v1, 0.0f); }
                if (out_bf16) {
                    unsigned short* d = (unsigned short*)Cout + (size_t)row * ldc + cOff + colBase;
                    d[0]  = f2bf(v0);
                    d[16] = f2bf(v1);
                } else {
                    float* d = (float*)Cout + (size_t)row * ldc + cOff + colBase;
                    d[0]  = v0;
                    d[16] = v1;
                }
            }
        }
    }
}

// ---------------------------------------------------------------------------
// Elementwise / graph / BN helpers
// ---------------------------------------------------------------------------
__global__ void fill_kernel(float* __restrict__ p, float v, long n) {
    long i = (long)blockIdx.x * blockDim.x + threadIdx.x;
    if (i < n) p[i] = v;
}

__global__ void deg_kernel(const long long* __restrict__ dst,
                           float* __restrict__ deg, int E) {
    int e = blockIdx.x * blockDim.x + threadIdx.x;
    if (e < E) atomicAdd(&deg[(int)dst[e]], 1.0f);
}

__global__ void norm_kernel(const float* __restrict__ deg,
                            float* __restrict__ dis,
                            float* __restrict__ invdeg, int n) {
    int i = blockIdx.x * blockDim.x + threadIdx.x;
    if (i < n) {
        float d = deg[i];
        dis[i]    = rsqrtf(d);
        invdeg[i] = 1.0f / d;
    }
}

// fp32 [N x F] -> bf16 slice cat[:, 0:F] with row stride ldo
__global__ void cvt_slice_kernel(const float* __restrict__ x,
                                 unsigned short* __restrict__ outp,
                                 int F, int ldo, long total) {
    long i = (long)blockIdx.x * blockDim.x + threadIdx.x;
    if (i < total) {
        long r = i / F; int c = (int)(i - r * F);
        outp[r * ldo + c] = f2bf(x[i]);
    }
}

// W [K x M] fp32 -> Wt [M x K] bf16  (one-time, tiny)
__global__ void wt_bf16_kernel(const float* __restrict__ W,
                               unsigned short* __restrict__ Wt, int K, int M) {
    long i = (long)blockIdx.x * blockDim.x + threadIdx.x;
    if (i < (long)K * M) {
        int m = (int)(i / K);
        int k = (int)(i - (long)m * K);
        Wt[i] = f2bf(W[(long)k * M + m]);
    }
}

// agg = xw * invdeg[row] + bias[col]   (GCN self-loop term, fp32)
__global__ void selfloop_kernel(const float* __restrict__ xw,
                                const float* __restrict__ invdeg,
                                const float* __restrict__ bias,
                                float* __restrict__ agg, int F, long total) {
    long i = (long)blockIdx.x * blockDim.x + threadIdx.x;
    if (i < total) {
        long r = i / F; int c = (int)(i - r * F);
        agg[i] = xw[i] * invdeg[r] + bias[c];
    }
}

// per-edge scatter: agg[dst] += xw[src] * dis[src]*dis[dst]   (block per edge)
__global__ void scatter_kernel(const long long* __restrict__ src,
                               const long long* __restrict__ dst,
                               const float* __restrict__ dis,
                               const float* __restrict__ xw,
                               float* __restrict__ agg, int F) {
    const int e = blockIdx.x;
    const int s = (int)src[e];
    const int d = (int)dst[e];
    const float coef = dis[s] * dis[d];
    const float* xs = xw  + (long)s * F;
    float*       ad = agg + (long)d * F;
    for (int f = threadIdx.x; f < F; f += blockDim.x)
        atomicAdd(&ad[f], xs[f] * coef);
}

__global__ void bn_stats_kernel(const float* __restrict__ x,
                                float* __restrict__ sum,
                                float* __restrict__ sumsq,
                                int F, int Nr) {
    int c = blockIdx.x * blockDim.x + threadIdx.x;
    if (c >= F) return;
    float s = 0.0f, s2 = 0.0f;
    for (int r = blockIdx.y; r < Nr; r += gridDim.y) {
        float v = x[(long)r * F + c];
        s += v; s2 += v * v;
    }
    atomicAdd(&sum[c],   s);
    atomicAdd(&sumsq[c], s2);
}

__global__ void bn_finalize_kernel(const float* __restrict__ sum,
                                   const float* __restrict__ sumsq,
                                   const float* __restrict__ g,
                                   const float* __restrict__ be,
                                   float* __restrict__ scale,
                                   float* __restrict__ shift,
                                   int F, float invN) {
    int c = blockIdx.x * blockDim.x + threadIdx.x;
    if (c >= F) return;
    float mean = sum[c] * invN;
    float var  = sumsq[c] * invN - mean * mean;
    float sc   = g[c] * rsqrtf(var + 1e-5f);
    scale[c] = sc;
    shift[c] = be[c] - mean * sc;
}

// y = relu(x*scale + shift) -> bf16 concat slice
__global__ void bn_apply_kernel(const float* __restrict__ x,
                                const float* __restrict__ scale,
                                const float* __restrict__ shift,
                                unsigned short* __restrict__ outp,
                                int F, int ldo, int cOff, long total) {
    long i = (long)blockIdx.x * blockDim.x + threadIdx.x;
    if (i < total) {
        long r = i / F; int c = (int)(i - r * F);
        float y = fmaxf(x[i] * scale[c] + shift[c], 0.0f);
        outp[r * ldo + cOff + c] = f2bf(y);
    }
}

// tiny-M dense layer (M = 10 / 3): bf16 A, fp32 weights/out
__global__ void small_gemm_kernel(const unsigned short* __restrict__ A, int K,
                                  const float* __restrict__ W,
                                  const float* __restrict__ bias,
                                  float* __restrict__ C, int M, int Nrows) {
    long i = (long)blockIdx.x * blockDim.x + threadIdx.x;
    long total = (long)Nrows * M;
    if (i >= total) return;
    long row = i / M; int col = (int)(i - row * M);
    const unsigned short* a = A + row * K;
    float acc = bias[col];
    #pragma unroll 4
    for (int k = 0; k < K; ++k) acc += bf2f(a[k]) * W[(long)k * M + col];
    C[i] = acc;
}

// ---------------------------------------------------------------------------
// Host orchestration
// ---------------------------------------------------------------------------
static inline int cdiv(long a, int b) { return (int)((a + b - 1) / b); }

static inline void launch_gemm(const unsigned short* A, int lda,
                               const unsigned short* Wt, int M, int K,
                               const float* bias, void* C, int ldc, int cOff,
                               int N, int out_bf16, int relu, hipStream_t s) {
    dim3 grid(M / 64, (unsigned)((N + 127) / 128));
    gemm_wmma<<<grid, 256, 0, s>>>(A, lda, Wt, K, bias, C, ldc, cOff, out_bf16,
                                   N, K, relu);
}

extern "C" void kernel_launch(void* const* d_in, const int* in_sizes, int n_in,
                              void* d_out, int out_size, void* d_ws, size_t ws_size,
                              hipStream_t stream) {
    (void)in_sizes; (void)n_in; (void)out_size; (void)ws_size;

    const float*     x_l   = (const float*)d_in[0];
    const float*     x_j   = (const float*)d_in[1];
    const long long* ei_l  = (const long long*)d_in[2];
    const long long* ei_j  = (const long long*)d_in[3];
    const float *lg_W1=(const float*)d_in[4],  *lg_b1=(const float*)d_in[5];
    const float *lg_g1=(const float*)d_in[6],  *lg_be1=(const float*)d_in[7];
    const float *lg_W2=(const float*)d_in[8],  *lg_b2=(const float*)d_in[9];
    const float *lg_g2=(const float*)d_in[10], *lg_be2=(const float*)d_in[11];
    const float *lg_fcW=(const float*)d_in[12],*lg_fcb=(const float*)d_in[13];
    const float *fl_W1=(const float*)d_in[14], *fl_b1=(const float*)d_in[15];
    const float *fl_W2=(const float*)d_in[16], *fl_b2=(const float*)d_in[17];
    const float *fl_W3=(const float*)d_in[18], *fl_b3=(const float*)d_in[19];
    const float *jg_W1=(const float*)d_in[20], *jg_b1=(const float*)d_in[21];
    const float *jg_g1=(const float*)d_in[22], *jg_be1=(const float*)d_in[23];
    const float *jg_W2=(const float*)d_in[24], *jg_b2=(const float*)d_in[25];
    const float *jg_g2=(const float*)d_in[26], *jg_be2=(const float*)d_in[27];
    const float *jg_fcW=(const float*)d_in[28],*jg_fcb=(const float*)d_in[29];
    const float *fj_W1=(const float*)d_in[30], *fj_b1=(const float*)d_in[31];
    const float *fj_W2=(const float*)d_in[32], *fj_b2=(const float*)d_in[33];

    float* out = (float*)d_out;
    float* ws  = (float*)d_ws;

    const int NL = 50000, NJ = 25000, EL = 400000, EJ = 100000;
    const int CL = 1792, CJ = 448;

    // workspace carve-out (float units; all chunks 16B-aligned)
    size_t off = 0;
    unsigned short* catLb = (unsigned short*)(ws + off); off += (size_t)NL * CL / 2;
    unsigned short* sc1b  = (unsigned short*)(ws + off); off += (size_t)NL * 1024 / 2;
    unsigned short* sc2b  = (unsigned short*)(ws + off); off += (size_t)NL * 1024 / 2;
    float*          xwL   = ws + off;                    off += (size_t)NL * 512;
    float*          aggL  = ws + off;                    off += (size_t)NL * 512;
    unsigned short* catJb = (unsigned short*)(ws + off); off += (size_t)NJ * CJ / 2;
    unsigned short* sj1b  = (unsigned short*)(ws + off); off += (size_t)NJ * 256 / 2;
    unsigned short* sj2b  = (unsigned short*)(ws + off); off += (size_t)NJ * 64 / 2;
    float*          xwJ   = ws + off;                    off += (size_t)NJ * 128;
    float*          aggJ  = ws + off;                    off += (size_t)NJ * 128;
    unsigned short* wLg1t = (unsigned short*)(ws + off); off += 512  * 1024 / 2;
    unsigned short* wLg2t = (unsigned short*)(ws + off); off += 256  * 512  / 2;
    unsigned short* wLfct = (unsigned short*)(ws + off); off += 1024 * 1792 / 2;
    unsigned short* wFl1t = (unsigned short*)(ws + off); off += 1024 * 1024 / 2;
    unsigned short* wFl2t = (unsigned short*)(ws + off); off += 1024 * 1024 / 2;
    unsigned short* wJg1t = (unsigned short*)(ws + off); off += 128 * 256 / 2;
    unsigned short* wJg2t = (unsigned short*)(ws + off); off += 64  * 128 / 2;
    unsigned short* wJfct = (unsigned short*)(ws + off); off += 256 * 448 / 2;
    unsigned short* wFj1t = (unsigned short*)(ws + off); off += 64  * 256 / 2;
    float* degL = ws + off; off += NL;
    float* disL = ws + off; off += NL;
    float* idgL = ws + off; off += NL;
    float* degJ = ws + off; off += NJ;
    float* disJ = ws + off; off += NJ;
    float* idgJ = ws + off; off += NJ;
    float* bsum = ws + off; off += 1024;
    float* bsq  = ws + off; off += 1024;
    float* bscl = ws + off; off += 1024;
    float* bsft = ws + off; off += 1024;

    // ---------------- one-time weight transpose+convert (bf16 [M][K]) -------
    auto wt = [&](const float* W, unsigned short* Wt, int K, int M) {
        wt_bf16_kernel<<<cdiv((long)K * M, 256), 256, 0, stream>>>(W, Wt, K, M);
    };
    wt(lg_W1, wLg1t, 1024, 512);  wt(lg_W2, wLg2t, 512, 256);
    wt(lg_fcW, wLfct, 1792, 1024);
    wt(fl_W1, wFl1t, 1024, 1024); wt(fl_W2, wFl2t, 1024, 1024);
    wt(jg_W1, wJg1t, 256, 128);   wt(jg_W2, wJg2t, 128, 64);
    wt(jg_fcW, wJfct, 448, 256);  wt(fj_W1, wFj1t, 256, 64);

    // ---------------- graph normalization ----------------
    fill_kernel<<<cdiv(NL,256),256,0,stream>>>(degL, 1.0f, NL);
    fill_kernel<<<cdiv(NJ,256),256,0,stream>>>(degJ, 1.0f, NJ);
    deg_kernel<<<cdiv(EL,256),256,0,stream>>>(ei_l + EL, degL, EL);
    deg_kernel<<<cdiv(EJ,256),256,0,stream>>>(ei_j + EJ, degJ, EJ);
    norm_kernel<<<cdiv(NL,256),256,0,stream>>>(degL, disL, idgL, NL);
    norm_kernel<<<cdiv(NJ,256),256,0,stream>>>(degJ, disJ, idgJ, NJ);

    // feats[0] = x -> bf16 concat slices
    cvt_slice_kernel<<<cdiv((long)NL*1024,256),256,0,stream>>>(x_l, catLb, 1024, CL, (long)NL*1024);
    cvt_slice_kernel<<<cdiv((long)NJ*256,256),256,0,stream>>>(x_j, catJb, 256, CJ, (long)NJ*256);

    // ---------------- LINE branch GCN layers ----------------
    struct Lay { const unsigned short *A; int lda, K, F;
                 const unsigned short *Wt; const float *b,*g,*be; int cOff; };
    Lay Llay[2] = {
        { catLb,        CL, 1024, 512, wLg1t, lg_b1, lg_g1, lg_be1, 1024 },
        { catLb + 1024, CL,  512, 256, wLg2t, lg_b2, lg_g2, lg_be2, 1536 },
    };
    for (int li = 0; li < 2; ++li) {
        const int F = Llay[li].F;
        const long tot = (long)NL * F;
        launch_gemm(Llay[li].A, Llay[li].lda, Llay[li].Wt, F, Llay[li].K,
                    nullptr, xwL, F, 0, NL, /*out_bf16=*/0, /*relu=*/0, stream);
        selfloop_kernel<<<cdiv(tot,256),256,0,stream>>>(xwL, idgL, Llay[li].b, aggL, F, tot);
        scatter_kernel<<<EL,256,0,stream>>>(ei_l, ei_l + EL, disL, xwL, aggL, F);
        fill_kernel<<<cdiv(F,256),256,0,stream>>>(bsum, 0.0f, F);
        fill_kernel<<<cdiv(F,256),256,0,stream>>>(bsq,  0.0f, F);
        bn_stats_kernel<<<dim3(cdiv(F,256),256),256,0,stream>>>(aggL, bsum, bsq, F, NL);
        bn_finalize_kernel<<<cdiv(F,256),256,0,stream>>>(bsum, bsq, Llay[li].g, Llay[li].be,
                                                         bscl, bsft, F, 1.0f/NL);
        bn_apply_kernel<<<cdiv(tot,256),256,0,stream>>>(aggL, bscl, bsft, catLb, F, CL,
                                                        Llay[li].cOff, tot);
    }
    // concat-FC + fc_lines chain (bf16 activations end-to-end)
    launch_gemm(catLb, CL,  wLfct, 1024, CL,   lg_fcb, sc1b, 1024, 0, NL, 1, 1, stream);
    launch_gemm(sc1b, 1024, wFl1t, 1024, 1024, fl_b1,  sc2b, 1024, 0, NL, 1, 1, stream);
    launch_gemm(sc2b, 1024, wFl2t, 1024, 1024, fl_b2,  sc1b, 1024, 0, NL, 1, 1, stream);
    small_gemm_kernel<<<cdiv((long)NL*10,256),256,0,stream>>>(sc1b, 1024, fl_W3, fl_b3, out, 10, NL);

    // ---------------- JUNCTION branch GCN layers ----------------
    Lay Jlay[2] = {
        { catJb,       CJ, 256, 128, wJg1t, jg_b1, jg_g1, jg_be1, 256 },
        { catJb + 256, CJ, 128,  64, wJg2t, jg_b2, jg_g2, jg_be2, 384 },
    };
    for (int li = 0; li < 2; ++li) {
        const int F = Jlay[li].F;
        const long tot = (long)NJ * F;
        launch_gemm(Jlay[li].A, Jlay[li].lda, Jlay[li].Wt, F, Jlay[li].K,
                    nullptr, xwJ, F, 0, NJ, 0, 0, stream);
        selfloop_kernel<<<cdiv(tot,256),256,0,stream>>>(xwJ, idgJ, Jlay[li].b, aggJ, F, tot);
        scatter_kernel<<<EJ,256,0,stream>>>(ei_j, ei_j + EJ, disJ, xwJ, aggJ, F);
        fill_kernel<<<cdiv(F,256),256,0,stream>>>(bsum, 0.0f, F);
        fill_kernel<<<cdiv(F,256),256,0,stream>>>(bsq,  0.0f, F);
        bn_stats_kernel<<<dim3(cdiv(F,256),256),256,0,stream>>>(aggJ, bsum, bsq, F, NJ);
        bn_finalize_kernel<<<cdiv(F,256),256,0,stream>>>(bsum, bsq, Jlay[li].g, Jlay[li].be,
                                                         bscl, bsft, F, 1.0f/NJ);
        bn_apply_kernel<<<cdiv(tot,256),256,0,stream>>>(aggJ, bscl, bsft, catJb, F, CJ,
                                                        Jlay[li].cOff, tot);
    }
    // concat-FC + fc_junction chain
    launch_gemm(catJb, CJ, wJfct, 256, CJ,  jg_fcb, sj1b, 256, 0, NJ, 1, 1, stream);
    launch_gemm(sj1b, 256, wFj1t, 64,  256, fj_b1,  sj2b, 64,  0, NJ, 1, 1, stream);
    small_gemm_kernel<<<cdiv((long)NJ*3,256),256,0,stream>>>(sj2b, 64, fj_W2, fj_b2,
                                                             out + (long)NL*10, 3, NJ);
}
